// FRFT_82729660056456
// MI455X (gfx1250) — matrix operand order repaired and checked by reading, not compiled, not run on hardware
//
#include <hip/hip_runtime.h>
#include <math.h>

// ---------------- problem constants ----------------
#define HWSZ   16384      // 128*128
#define NBATCH 4
#define CIN    192
#define CH1    48         // C1
#define CH0    16         // C0
#define CHM    16         // Cm
#define COUT   192
#define NPIX   (NBATCH*HWSZ)   // 65536
#define NK     65         // rfft bins
#define PKW    (128*NK)   // 8320 per (n,c) plane in freq domain

typedef __attribute__((ext_vector_type(16))) _Float16 v16h;
typedef __attribute__((ext_vector_type(8)))  float    v8f;
typedef _Float16 half_t;

#define PI_F 3.14159265358979f

// ---------------- async global->LDS helpers (CDNA5, ASYNCcnt tracked) ----------------
__device__ inline void async_g2l_b32(void* lds, const void* g) {
  unsigned l32 = (unsigned)(size_t)lds;                 // LDS offset = low 32 bits of generic ptr
  unsigned long long ga = (unsigned long long)(size_t)g;
  asm volatile("global_load_async_to_lds_b32 %0, %1, off" :: "v"(l32), "v"(ga) : "memory");
}
__device__ inline void wait_async0() {
  asm volatile("s_wait_asynccnt 0" ::: "memory");
}

// ---------------- WMMA fragment helpers (CDNA5 16x16x32 f16) ----------------
// A-matrix 16x32 layout (ISA 7.12.2): lane L holds row M=L%16; half=L/16.
// VGPR v<4 -> K = 2v + 8*half + {0,1}; v>=4 -> K = 16 + 2(v-4) + 8*half + {0,1}
__device__ inline v16h load_a_frag(const half_t* t, int ld) {
  int lane = threadIdx.x & 31;
  int m = lane & 15, hi = lane >> 4;
  v16h a;
#pragma unroll
  for (int e = 0; e < 16; ++e) {
    int v = e >> 1, odd = e & 1;
    int k = ((v < 4) ? (2 * v) : (16 + 2 * (v - 4))) + 8 * hi + odd;
    a[e] = t[m * ld + k];
  }
  return a;
}
// ---- packed B layout: per (k-chunk ck of 32, n-tile nt of 16, lane) the 16
// fragment halves are contiguous: index = ((ck*8+nt)*32+lane)*16 + e,
// where element (K=kc+32*ck, N=j) maps to lane=(((kc>>3)&1)<<4)|(j&15), e=(kc&7)+8*(kc>>4).
__device__ inline int pk_read(int ck, int nt, int lane) {
  return ((ck * 8 + nt) * 32 + lane) * 16;
}
__device__ inline int pk_write(int kc, int ck, int j) {
  int e = (kc & 7) + 8 * (kc >> 4), hi = (kc >> 3) & 1;
  int l = (hi << 4) | (j & 15);
  return ((ck * 8 + (j >> 4)) * 32 + l) * 16 + e;
}
__device__ inline v16h load_b_packed(const half_t* base, int ck, int nt) {
  int lane = threadIdx.x & 31;
  return *(const v16h*)(base + pk_read(ck, nt, lane));   // 32B aligned -> 2x b128
}
__device__ inline v8f wmma16(v16h a, v16h b, v8f c) {
  return __builtin_amdgcn_wmma_f32_16x16x32_f16(false, a, false, b, (short)0, c, false, false);
}

// ---------------- 1. build FRFT matrices (A = chirp-FRFT op, C = centered ifft) ----------------
__global__ void k_build_mats(float* A_re, float* A_im, float* C_re, float* C_im) {
  int idx = blockIdx.x * blockDim.x + threadIdx.x;
  if (idx >= 128 * 128) return;
  int k = idx >> 7, t = idx & 127;
  const double PI = 3.14159265358979323846;
  double tana2 = tan(PI / 8.0);                 // tan(alpha/2), alpha=pi/4
  double sina  = sin(PI / 4.0);
  double cc    = PI / (128.0 * sina * 4.0);     // chirp-conv coefficient
  double chco  = PI / 128.0 * tana2 / 4.0;      // chrp coefficient
  int s_out = 127 + 2 * k;                      // decimated output index
  double sr = 0.0, si = 0.0;
  for (int r = 127; r <= 381; ++r) {            // nonzero fp region
    int m = r - 127;
    double sterm;
    if ((m & 1) == 0) {                         // even: sinc delta (sample preserved)
      if (m != 2 * t) continue;
      sterm = 1.0;
    } else {                                    // odd: sinc((m-2t)/2)
      int k0 = m - 2 * t;                       // odd, |k0|<=253
      int km = ((k0 % 4) + 4) % 4;
      double sgn = (km == 1) ? 1.0 : -1.0;      // sin(pi*k0/2)
      sterm = sgn * 2.0 / (PI * (double)k0);
    }
    double n1 = (double)(r - 254);
    double d  = (double)(s_out - r);
    double ang = -chco * n1 * n1 + cc * d * d;  // chrp[r] * kern2[s-r]
    sr += sterm * cos(ang);
    si += sterm * sin(ang);
  }
  // phi = exp(-i pi/8) * chrp[s_out] * sqrt(cc/pi)
  double n2 = (double)(s_out - 254);
  double ak = -chco * n2 * n2 - PI / 8.0;
  double sc = sqrt(cc / PI);
  double pr = cos(ak) * sc, pi2 = sin(ak) * sc;
  A_re[idx] = (float)(sr * pr - si * pi2);
  A_im[idx] = (float)(sr * pi2 + si * pr);
  // centered-ifft matrix: (1/sqrt(128)) exp(2 pi i (k+64)(t+64)/128)
  int mm = ((k + 64) * (t + 64)) & 127;
  double angc = 2.0 * PI * (double)mm / 128.0;
  double inv = 1.0 / sqrt(128.0);
  C_re[idx] = (float)(cos(angc) * inv);
  C_im[idx] = (float)(sin(angc) * inv);
}

// ---------------- 2. B = A*C; write WMMA-packed f16 operand images ----------------
// Element M[k][t] is B-operand entry (K=t, N=k); packed for fragment-contiguous loads.
__global__ void k_compose(const float* A_re, const float* A_im,
                          const float* C_re, const float* C_im,
                          float* B_re, float* B_im,
                          half_t* PA_re, half_t* PA_im, half_t* PA_imn,
                          half_t* PB_re, half_t* PB_im, half_t* PB_imn) {
  int idx = blockIdx.x * blockDim.x + threadIdx.x;
  if (idx >= 128 * 128) return;
  int k = idx >> 7, t = idx & 127;
  float br = 0.f, bi = 0.f;
  for (int u = 0; u < 128; ++u) {
    float ar = A_re[k * 128 + u], ai = A_im[k * 128 + u];
    float cr = C_re[u * 128 + t], ci = C_im[u * 128 + t];
    br += ar * cr - ai * ci;
    bi += ar * ci + ai * cr;
  }
  B_re[idx] = br; B_im[idx] = bi;
  int pk = pk_write(t & 31, t >> 5, k);
  float are = A_re[idx], aim = A_im[idx];
  PA_re[pk] = (half_t)are; PA_im[pk] = (half_t)aim; PA_imn[pk] = (half_t)(-aim);
  PB_re[pk] = (half_t)br;  PB_im[pk] = (half_t)bi;  PB_imn[pk] = (half_t)(-bi);
}

// ---------------- 3. convert conv weights to f16 (conv2 K padded 48->64) ----------------
__global__ void k_convert_weights(const float* w1, const float* w2, half_t* w1h, half_t* w2h) {
  int i = blockIdx.x * blockDim.x + threadIdx.x;
  if (i < CH1 * CIN) w1h[i] = (half_t)w1[i];
  if (i < COUT * 64) {
    int o = i >> 6, c = i & 63;
    w2h[i] = (c < CH1) ? (half_t)w2[o * CH1 + c] : (half_t)0.0f;
  }
}

// ---------------- 4. conv1 (1x1, 192->48) via WMMA ----------------
__global__ __launch_bounds__(256) void k_conv1(const float* __restrict__ x,
                                               const half_t* __restrict__ w1h,
                                               float* __restrict__ x1) {
  __shared__ __align__(32) half_t sX[CIN * 128];   // packed B layout (6 k-chunks x 8 n-tiles)
  __shared__ __align__(32) half_t sW[16 * CIN];    // 16 out rows x 192 (A operand)
  int tid = threadIdx.x;
  int mt = blockIdx.y;                  // 0..2 (M tiles of 16)
  int colBase = blockIdx.x * 128;
  __builtin_prefetch(x + (size_t)(colBase + 128), 0, 1);   // global_prefetch_b8
  for (int i = tid; i < 16 * CIN; i += 256) {
    int o = i / CIN, c = i - o * CIN;
    sW[i] = w1h[(mt * 16 + o) * CIN + c];
  }
  for (int i = tid; i < CIN * 128; i += 256) {
    int c = i >> 7, j = i & 127;
    int q = colBase + j;
    int n = q >> 14, p = q & (HWSZ - 1);
    sX[pk_write(c & 31, c >> 5, j)] = (half_t)x[((size_t)n * CIN + c) * HWSZ + p];
  }
  __syncthreads();
  int wv = tid >> 5, lane = tid & 31;
  int ncol = lane & 15, hi = lane >> 4;
  v8f acc = {};
#pragma unroll
  for (int ck = 0; ck < CIN / 32; ++ck) {
    v16h a = load_a_frag(sW + ck * 32, CIN);
    v16h b = load_b_packed(sX, ck, wv);
    acc = wmma16(a, b, acc);
  }
  int q = colBase + wv * 16 + ncol;
  int n = q >> 14, p = q & (HWSZ - 1);
#pragma unroll
  for (int v = 0; v < 8; ++v) {
    int o = mt * 16 + v + 8 * hi;
    x1[((size_t)n * CH1 + o) * HWSZ + p] = acc[v];
  }
}

// ---------------- 5. extract x_05 slice (real -> complex buffers) ----------------
__global__ void k_extract05(const float* __restrict__ x1, float* t_re, float* t_im) {
  size_t i = (size_t)blockIdx.x * blockDim.x + threadIdx.x;
  if (i >= (size_t)NBATCH * CHM * HWSZ) return;
  int p = (int)(i & (HWSZ - 1));
  int c = (int)((i >> 14) & 15);
  int n = (int)(i >> 18);
  t_re[i] = x1[((size_t)n * CH1 + CH0 + c) * HWSZ + p];
  t_im[i] = 0.f;
}

// ---------------- 6. complex rows x 128x128 matrix via WMMA; inIm may be null ----------------
__global__ __launch_bounds__(256) void k_rowmm(const float* __restrict__ inRe,
                                               const float* __restrict__ inIm,
                                               const half_t* __restrict__ PRe,
                                               const half_t* __restrict__ PIm,
                                               const half_t* __restrict__ PImN,
                                               float* __restrict__ outRe,
                                               float* __restrict__ outIm) {
  __shared__ __align__(32) half_t sAre[16 * 128];
  __shared__ __align__(32) half_t sAim[16 * 128];
  int tid = threadIdx.x;
  size_t rowBase = (size_t)blockIdx.x * 16;
  for (int i = tid; i < 16 * 128; i += 256) {
    int r = i >> 7, t = i & 127;
    size_t g = (rowBase + r) * 128 + t;
    sAre[i] = (half_t)inRe[g];
    sAim[i] = inIm ? (half_t)inIm[g] : (half_t)0.0f;
  }
  __syncthreads();
  int wv = tid >> 5, lane = tid & 31;
  int ncol = lane & 15, hi = lane >> 4;
  v8f accR = {}, accI = {};
#pragma unroll
  for (int ck = 0; ck < 4; ++ck) {
    v16h aR = load_a_frag(sAre + ck * 32, 128);
    v16h aI = load_a_frag(sAim + ck * 32, 128);
    v16h bR  = load_b_packed(PRe, ck, wv);
    v16h bI  = load_b_packed(PIm, ck, wv);
    v16h bIN = load_b_packed(PImN, ck, wv);
    accR = wmma16(aR, bR, accR);
    accR = wmma16(aI, bIN, accR);   // Re -= Ai*Bi (pre-negated B)
    accI = wmma16(aR, bI, accI);
    accI = wmma16(aI, bR, accI);
  }
#pragma unroll
  for (int v = 0; v < 8; ++v) {
    size_t g = (rowBase + v + 8 * hi) * 128 + (wv * 16 + ncol);
    outRe[g] = accR[v];
    outIm[g] = accI[v];
  }
}

// ---------------- 7. per-image 128x128 complex transpose ----------------
__global__ __launch_bounds__(256) void k_transpose(const float* in_re, const float* in_im,
                                                   float* out_re, float* out_im) {
  __shared__ float tile[32][33];
  int img = blockIdx.z;
  int bx = blockIdx.x, by = blockIdx.y;
  size_t base = (size_t)img * HWSZ;
  int tx = threadIdx.x & 31, tg = threadIdx.x >> 5;
  for (int plane = 0; plane < 2; ++plane) {
    const float* in = plane ? in_im : in_re;
    float* out = plane ? out_im : out_re;
    __syncthreads();
    for (int r = tg; r < 32; r += 8)
      tile[r][tx] = in[base + (size_t)(by * 32 + r) * 128 + bx * 32 + tx];
    __syncthreads();
    for (int r = tg; r < 32; r += 8)
      out[base + (size_t)(bx * 32 + r) * 128 + by * 32 + tx] = tile[tx][r];
  }
}

// ---------------- 8. mag / phase / mag*mask1 ----------------
__global__ void k_magphase(const float* re_, const float* im_,
                           float* mag, float* pha, float* magM1) {
  size_t i = (size_t)blockIdx.x * blockDim.x + threadIdx.x;
  if (i >= (size_t)NBATCH * CHM * HWSZ) return;
  float re = re_[i], im = im_[i];
  float m = sqrtf(re * re + im * im);
  pha[i] = atan2f(im, re);
  mag[i] = m;
  int pix = (int)(i & (HWSZ - 1));
  int h = pix >> 7, w = pix & 127;
  bool in1 = (h >= 19 && h < 109 && w >= 19 && w < 109);
  magM1[i] = in1 ? m : 0.f;
}

// ---------------- 9. mag_f branch: 1x1 on mag*mask2, *mask2 -> init magmix ----------------
__global__ __launch_bounds__(256) void k_magf(const float* __restrict__ mag,
                                              const float* __restrict__ wf,
                                              const float* __restrict__ bf,
                                              float* __restrict__ magmix) {
  __shared__ float sw[CHM * CHM];
  __shared__ float sb[CHM];
  int tid = threadIdx.x;
  if (tid < CHM * CHM) sw[tid] = wf[tid];
  if (tid < CHM) sb[tid] = bf[tid];
  __syncthreads();
  size_t i = (size_t)blockIdx.x * blockDim.x + tid;
  if (i >= (size_t)NBATCH * CHM * HWSZ) return;
  int pix = (int)(i & (HWSZ - 1));
  int o = (int)((i >> 14) & 15);
  int n = (int)(i >> 18);
  int h = pix >> 7, w = pix & 127;
  bool in1 = (h >= 19 && h < 109 && w >= 19 && w < 109);
  float val = 0.f;
  if (!in1) {
    float acc = sb[o];
    for (int c = 0; c < CHM; ++c)
      acc += sw[o * CHM + c] * mag[((size_t)n * CHM + c) * HWSZ + pix];
    val = acc;
  }
  magmix[i] = val;
}

// ---------------- 10. mag_s branch: 3x3 SAME on mag*mask1, +bias, *mask1, += magmix ----------------
__global__ __launch_bounds__(256) void k_mags3(const float* __restrict__ magM1,
                                               const float* __restrict__ ws,
                                               const float* __restrict__ bs,
                                               float* __restrict__ magmix) {
  __shared__ float sw[CHM * CHM * 9];
  __shared__ float sb[CHM];
  int tid = threadIdx.x;
  for (int i = tid; i < CHM * CHM * 9; i += 256) sw[i] = ws[i];
  if (tid < CHM) sb[tid] = bs[tid];
  __syncthreads();
  size_t i = (size_t)blockIdx.x * blockDim.x + tid;
  if (i >= (size_t)NBATCH * CHM * HWSZ) return;
  int pix = (int)(i & (HWSZ - 1));
  int o = (int)((i >> 14) & 15);
  int n = (int)(i >> 18);
  int h = pix >> 7, w = pix & 127;
  bool in1 = (h >= 19 && h < 109 && w >= 19 && w < 109);
  if (!in1) return;
  float acc = sb[o];
  for (int c = 0; c < CHM; ++c) {
    const float* pl = magM1 + ((size_t)n * CHM + c) * HWSZ;
    const float* wk = sw + (o * CHM + c) * 9;
#pragma unroll
    for (int ky = 0; ky < 3; ++ky) {
      int hh = h + ky - 1;
      if (hh < 0 || hh > 127) continue;
#pragma unroll
      for (int kx = 0; kx < 3; ++kx) {
        int ww = w + kx - 1;
        if (ww < 0 || ww > 127) continue;
        acc += wk[ky * 3 + kx] * pl[hh * 128 + ww];
      }
    }
  }
  magmix[i] += acc;
}

// ---------------- 11. mag_out / pha_out 1x1s -> Fre_out complex ----------------
__global__ __launch_bounds__(256) void k_freout(const float* __restrict__ magmix,
                                                const float* __restrict__ pha,
                                                const float* __restrict__ magw,
                                                const float* __restrict__ magb,
                                                const float* __restrict__ phaw,
                                                const float* __restrict__ phab,
                                                float* __restrict__ o_re,
                                                float* __restrict__ o_im) {
  __shared__ float swm[CHM * CHM], swp[CHM * CHM], sbm[CHM], sbp[CHM];
  int tid = threadIdx.x;
  if (tid < CHM * CHM) { swm[tid] = magw[tid]; swp[tid] = phaw[tid]; }
  if (tid < CHM) { sbm[tid] = magb[tid]; sbp[tid] = phab[tid]; }
  __syncthreads();
  size_t i = (size_t)blockIdx.x * blockDim.x + tid;
  if (i >= (size_t)NBATCH * CHM * HWSZ) return;
  int pix = (int)(i & (HWSZ - 1));
  int o = (int)((i >> 14) & 15);
  int n = (int)(i >> 18);
  float am = sbm[o], ap = sbp[o];
  for (int c = 0; c < CHM; ++c) {
    size_t idx = ((size_t)n * CHM + c) * HWSZ + pix;
    am += swm[o * CHM + c] * magmix[idx];
    ap += swp[o * CHM + c] * pha[idx];
  }
  float sn, cs;
  sincosf(ap, &sn, &cs);
  o_re[i] = am * cs;
  o_im[i] = am * sn;
}

// ---------------- 12. |.| of inverse FRFT -> cat channels 16..31 ----------------
__global__ void k_abscat(const float* re_, const float* im_, float* cat) {
  size_t i = (size_t)blockIdx.x * blockDim.x + threadIdx.x;
  if (i >= (size_t)NBATCH * CHM * HWSZ) return;
  int pix = (int)(i & (HWSZ - 1));
  int c = (int)((i >> 14) & 15);
  int n = (int)(i >> 18);
  float re = re_[i], im = im_[i];
  cat[((size_t)n * CH1 + CH0 + c) * HWSZ + pix] = sqrtf(re * re + im * im);
}

// ---------------- 13. x_0 branch 1x1 -> cat channels 0..15 ----------------
__global__ __launch_bounds__(256) void k_conv0(const float* __restrict__ x1,
                                               const float* __restrict__ w,
                                               const float* __restrict__ b,
                                               float* __restrict__ cat) {
  __shared__ float sw[CH0 * CH0], sb[CH0];
  int tid = threadIdx.x;
  if (tid < CH0 * CH0) sw[tid] = w[tid];
  if (tid < CH0) sb[tid] = b[tid];
  __syncthreads();
  size_t i = (size_t)blockIdx.x * blockDim.x + tid;
  if (i >= (size_t)NBATCH * CH0 * HWSZ) return;
  int pix = (int)(i & (HWSZ - 1));
  int o = (int)((i >> 14) & 15);
  int n = (int)(i >> 18);
  float acc = sb[o];
  for (int c = 0; c < CH0; ++c)
    acc += sw[o * CH0 + c] * x1[((size_t)n * CH1 + c) * HWSZ + pix];
  cat[((size_t)n * CH1 + o) * HWSZ + pix] = acc;
}

// ---------------- 14. rfft along W (async LDS staging) ----------------
__global__ __launch_bounds__(128) void k_rfftW(const float* __restrict__ x1,
                                               float* __restrict__ f_re, float* __restrict__ f_im) {
  __shared__ float row[128];
  __shared__ float ctab[128], stab[128];
  int rid = blockIdx.x;                 // (n,c,h)
  int h = rid & 127, c = (rid >> 7) & 15, n = rid >> 11;
  int t = threadIdx.x;
  float a = 2.f * PI_F * (float)t / 128.f;
  ctab[t] = cosf(a); stab[t] = sinf(a);
  async_g2l_b32(&row[t], &x1[(((size_t)n * CH1 + 32 + c) * 128 + h) * 128 + t]);
  wait_async0();
  __syncthreads();
  if (t < NK) {
    float sr = 0.f, si = 0.f;
    for (int w = 0; w < 128; ++w) {
      int id = (w * t) & 127;
      sr += row[w] * ctab[id];
      si -= row[w] * stab[id];
    }
    size_t o = (size_t)rid * NK + t;
    f_re[o] = sr; f_im[o] = si;
  }
}

// ---------------- 15. fft along H (complex columns, async staging) ----------------
__global__ __launch_bounds__(128) void k_fftH(const float* __restrict__ in_re,
                                              const float* __restrict__ in_im,
                                              float* __restrict__ o_re, float* __restrict__ o_im) {
  __shared__ float cre[128], cim[128];
  __shared__ float ctab[128], stab[128];
  int col = blockIdx.x;                 // (n*16+c)*65 + k
  int k = col % NK, nc = col / NK;
  int t = threadIdx.x;
  float a = 2.f * PI_F * (float)t / 128.f;
  ctab[t] = cosf(a); stab[t] = sinf(a);
  size_t base = (size_t)nc * 128 * NK + k;
  async_g2l_b32(&cre[t], &in_re[base + (size_t)t * NK]);
  async_g2l_b32(&cim[t], &in_im[base + (size_t)t * NK]);
  wait_async0();
  __syncthreads();
  float sr = 0.f, si = 0.f;
  for (int j = 0; j < 128; ++j) {
    int id = (t * j) & 127;
    float c0 = ctab[id], s0 = -stab[id];
    sr += cre[j] * c0 - cim[j] * s0;
    si += cre[j] * s0 + cim[j] * c0;
  }
  o_re[base + (size_t)t * NK] = sr;
  o_im[base + (size_t)t * NK] = si;
}

// ---------------- 16. mag/phase of fre ----------------
__global__ void k_mpfft(const float* z_re, const float* z_im, float* fm, float* fp) {
  size_t i = (size_t)blockIdx.x * blockDim.x + threadIdx.x;
  if (i >= (size_t)NBATCH * CHM * PKW) return;
  float re = z_re[i], im = z_im[i];
  fm[i] = sqrtf(re * re + im * im);
  fp[i] = atan2f(im, re);
}

// ---------------- 17. conv_1 1x1 on mag & phase -> Z complex ----------------
__global__ __launch_bounds__(256) void k_zfft(const float* __restrict__ fm,
                                              const float* __restrict__ fp,
                                              const float* __restrict__ w,
                                              const float* __restrict__ b,
                                              float* __restrict__ z_re, float* __restrict__ z_im) {
  __shared__ float sw[CH0 * CH0], sb[CH0];
  int tid = threadIdx.x;
  if (tid < CH0 * CH0) sw[tid] = w[tid];
  if (tid < CH0) sb[tid] = b[tid];
  __syncthreads();
  size_t i = (size_t)blockIdx.x * blockDim.x + tid;
  if (i >= (size_t)NBATCH * CHM * PKW) return;
  int pk = (int)(i % PKW);
  int o = (int)((i / PKW) % 16);
  int n = (int)(i / ((size_t)16 * PKW));
  float am = sb[o], ap = sb[o];
  for (int c = 0; c < 16; ++c) {
    size_t idx = ((size_t)(n * 16 + c)) * PKW + pk;
    am += sw[o * 16 + c] * fm[idx];
    ap += sw[o * 16 + c] * fp[idx];
  }
  float sn, cs;
  sincosf(ap, &sn, &cs);
  z_re[i] = am * cs;
  z_im[i] = am * sn;
}

// ---------------- 18. ifft along H (1/128, async staging) ----------------
__global__ __launch_bounds__(128) void k_ifftH(const float* __restrict__ in_re,
                                               const float* __restrict__ in_im,
                                               float* __restrict__ o_re, float* __restrict__ o_im) {
  __shared__ float cre[128], cim[128];
  __shared__ float ctab[128], stab[128];
  int col = blockIdx.x;
  int k = col % NK, nc = col / NK;
  int t = threadIdx.x;
  float a = 2.f * PI_F * (float)t / 128.f;
  ctab[t] = cosf(a); stab[t] = sinf(a);
  size_t base = (size_t)nc * 128 * NK + k;
  async_g2l_b32(&cre[t], &in_re[base + (size_t)t * NK]);
  async_g2l_b32(&cim[t], &in_im[base + (size_t)t * NK]);
  wait_async0();
  __syncthreads();
  float sr = 0.f, si = 0.f;
  for (int j = 0; j < 128; ++j) {
    int id = (t * j) & 127;
    float c0 = ctab[id], s0 = stab[id];
    sr += cre[j] * c0 - cim[j] * s0;
    si += cre[j] * s0 + cim[j] * c0;
  }
  o_re[base + (size_t)t * NK] = sr * (1.f / 128.f);
  o_im[base + (size_t)t * NK] = si * (1.f / 128.f);
}

// ---------------- 19. irfft along W -> cat channels 32..47 ----------------
__global__ __launch_bounds__(128) void k_irfftW(const float* __restrict__ t_re,
                                                const float* __restrict__ t_im,
                                                float* __restrict__ cat) {
  __shared__ float zr[NK], zi[NK];
  __shared__ float ctab[128], stab[128];
  int rid = blockIdx.x;
  int h = rid & 127, c = (rid >> 7) & 15, n = rid >> 11;
  int t = threadIdx.x;
  float a = 2.f * PI_F * (float)t / 128.f;
  ctab[t] = cosf(a); stab[t] = sinf(a);
  size_t base = (size_t)rid * NK;
  if (t < NK) {
    async_g2l_b32(&zr[t], &t_re[base + t]);
    async_g2l_b32(&zi[t], &t_im[base + t]);
  }
  wait_async0();
  __syncthreads();
  int w = t;
  float acc = zr[0] + ((w & 1) ? -zr[64] : zr[64]);
  for (int k = 1; k < 64; ++k) {
    int id = (w * k) & 127;
    acc += 2.f * (zr[k] * ctab[id] - zi[k] * stab[id]);
  }
  acc *= (1.f / 128.f);
  cat[(((size_t)n * CH1 + 32 + c) * 128 + h) * 128 + w] = acc;
}

// ---------------- 20. conv2 (1x1, 48->192) via WMMA (K padded to 64) ----------------
__global__ __launch_bounds__(256) void k_conv2(const float* __restrict__ cat,
                                               const half_t* __restrict__ w2h,
                                               float* __restrict__ out) {
  __shared__ __align__(32) half_t sX[64 * 128];   // packed B layout (2 k-chunks x 8 n-tiles)
  int tid = threadIdx.x;
  int colBase = blockIdx.x * 128;
  for (int i = tid; i < 64 * 128; i += 256) {
    int c = i >> 7, j = i & 127;
    int q = colBase + j;
    int n = q >> 14, p = q & (HWSZ - 1);
    float v = (c < CH1) ? cat[((size_t)n * CH1 + c) * HWSZ + p] : 0.f;
    sX[pk_write(c & 31, c >> 5, j)] = (half_t)v;
  }
  __syncthreads();
  int wv = tid >> 5, lane = tid & 31;
  int ncol = lane & 15, hi = lane >> 4;
  for (int mt = 0; mt < 12; ++mt) {
    v8f acc = {};
#pragma unroll
    for (int ck = 0; ck < 2; ++ck) {
      v16h a = load_a_frag(w2h + (size_t)(mt * 16) * 64 + ck * 32, 64);
      v16h b = load_b_packed(sX, ck, wv);
      acc = wmma16(a, b, acc);
    }
    int q = colBase + wv * 16 + ncol;
    int n = q >> 14, p = q & (HWSZ - 1);
#pragma unroll
    for (int v = 0; v < 8; ++v) {
      int o = mt * 16 + v + 8 * hi;
      out[((size_t)n * COUT + o) * HWSZ + p] = acc[v];
    }
  }
}

// ---------------- host launch ----------------
extern "C" void kernel_launch(void* const* d_in, const int* in_sizes, int n_in,
                              void* d_out, int out_size, void* d_ws, size_t ws_size,
                              hipStream_t stream) {
  const float* x       = (const float*)d_in[0];
  const float* conv1_w = (const float*)d_in[1];
  const float* mag_s_w = (const float*)d_in[2];
  const float* mag_s_b = (const float*)d_in[3];
  const float* mag_f_w = (const float*)d_in[4];
  const float* mag_f_b = (const float*)d_in[5];
  const float* mag_w   = (const float*)d_in[6];
  const float* mag_b   = (const float*)d_in[7];
  const float* pha_w   = (const float*)d_in[8];
  const float* pha_b   = (const float*)d_in[9];
  const float* conv_0_w = (const float*)d_in[10];
  const float* conv_0_b = (const float*)d_in[11];
  const float* conv_1_w = (const float*)d_in[12];
  const float* conv_1_b = (const float*)d_in[13];
  const float* conv2_w  = (const float*)d_in[14];
  float* out = (float*)d_out;

  // workspace layout (f32 region then f16 region)
  const size_t nPlane = (size_t)NBATCH * CHM * HWSZ;   // 1,048,576
  const size_t nFreq  = (size_t)NBATCH * CHM * PKW;    // 532,480
  float* W = (float*)d_ws;
  size_t off = 0;
  float* A_re = W + off; off += 16384;
  float* A_im = W + off; off += 16384;
  float* C_re = W + off; off += 16384;
  float* C_im = W + off; off += 16384;
  float* B_re = W + off; off += 16384;
  float* B_im = W + off; off += 16384;
  float* x1   = W + off; off += (size_t)NBATCH * CH1 * HWSZ;
  float* t1_re = W + off; off += nPlane;
  float* t1_im = W + off; off += nPlane;
  float* t2_re = W + off; off += nPlane;
  float* t2_im = W + off; off += nPlane;
  float* mag  = W + off; off += nPlane;
  float* pha  = W + off; off += nPlane;
  float* f1_re = W + off; off += nFreq;
  float* f1_im = W + off; off += nFreq;
  float* f2_re = W + off; off += nFreq;
  float* f2_im = W + off; off += nFreq;
  float* cat  = W + off; off += (size_t)NBATCH * CH1 * HWSZ;
  half_t* HB = (half_t*)(W + off);
  size_t hoff = 0;
  half_t* PA_re = HB + hoff; hoff += 16384;
  half_t* PA_im = HB + hoff; hoff += 16384;
  half_t* PA_in = HB + hoff; hoff += 16384;
  half_t* PB_re = HB + hoff; hoff += 16384;
  half_t* PB_im = HB + hoff; hoff += 16384;
  half_t* PB_in = HB + hoff; hoff += 16384;
  half_t* w1h = HB + hoff; hoff += (size_t)CH1 * CIN;
  half_t* w2h = HB + hoff; hoff += (size_t)COUT * 64;
  size_t needed = off * sizeof(float) + hoff * sizeof(half_t);
  if (ws_size < needed) return;

  const int TB = 256;
  // setup: transform matrices + weight conversion
  k_build_mats<<<64, TB, 0, stream>>>(A_re, A_im, C_re, C_im);
  k_compose<<<64, TB, 0, stream>>>(A_re, A_im, C_re, C_im, B_re, B_im,
                                   PA_re, PA_im, PA_in, PB_re, PB_im, PB_in);
  k_convert_weights<<<(COUT * 64 + TB - 1) / TB, TB, 0, stream>>>(conv1_w, conv2_w, w1h, w2h);

  // conv1 (WMMA)
  k_conv1<<<dim3(NPIX / 128, 3), TB, 0, stream>>>(x, w1h, x1);

  // forward FRFT2D on x_05 : rows*A, T, rows*A, T  -> Fre in t1
  int nBlkP = (int)(nPlane / TB);
  k_extract05<<<nBlkP, TB, 0, stream>>>(x1, t1_re, t1_im);
  k_rowmm<<<512, TB, 0, stream>>>(t1_re, (const float*)nullptr, PA_re, PA_im, PA_in, t2_re, t2_im);
  k_transpose<<<dim3(4, 4, NBATCH * CHM), TB, 0, stream>>>(t2_re, t2_im, t1_re, t1_im);
  k_rowmm<<<512, TB, 0, stream>>>(t1_re, t1_im, PA_re, PA_im, PA_in, t2_re, t2_im);
  k_transpose<<<dim3(4, 4, NBATCH * CHM), TB, 0, stream>>>(t2_re, t2_im, t1_re, t1_im);

  // mag/phase processing; magM1 -> t2_re, magmix -> t2_im
  k_magphase<<<nBlkP, TB, 0, stream>>>(t1_re, t1_im, mag, pha, t2_re);
  k_magf<<<nBlkP, TB, 0, stream>>>(mag, mag_f_w, mag_f_b, t2_im);
  k_mags3<<<nBlkP, TB, 0, stream>>>(t2_re, mag_s_w, mag_s_b, t2_im);
  k_freout<<<nBlkP, TB, 0, stream>>>(t2_im, pha, mag_w, mag_b, pha_w, pha_b, t1_re, t1_im);

  // inverse FRFT2D : rows*B, T, rows*B, T ; |.| -> cat ch16..31
  k_rowmm<<<512, TB, 0, stream>>>(t1_re, t1_im, PB_re, PB_im, PB_in, t2_re, t2_im);
  k_transpose<<<dim3(4, 4, NBATCH * CHM), TB, 0, stream>>>(t2_re, t2_im, t1_re, t1_im);
  k_rowmm<<<512, TB, 0, stream>>>(t1_re, t1_im, PB_re, PB_im, PB_in, t2_re, t2_im);
  k_transpose<<<dim3(4, 4, NBATCH * CHM), TB, 0, stream>>>(t2_re, t2_im, t1_re, t1_im);
  k_abscat<<<nBlkP, TB, 0, stream>>>(t1_re, t1_im, cat);

  // x_0 branch -> cat ch0..15
  k_conv0<<<nBlkP, TB, 0, stream>>>(x1, conv_0_w, conv_0_b, cat);

  // rfft2 branch on x_1 -> cat ch32..47
  int nBlkF = (int)((nFreq + TB - 1) / TB);
  k_rfftW<<<NBATCH * CHM * 128, 128, 0, stream>>>(x1, f1_re, f1_im);
  k_fftH<<<NBATCH * CHM * NK, 128, 0, stream>>>(f1_re, f1_im, f2_re, f2_im);
  k_mpfft<<<nBlkF, TB, 0, stream>>>(f2_re, f2_im, f1_re, f1_im);
  k_zfft<<<nBlkF, TB, 0, stream>>>(f1_re, f1_im, conv_1_w, conv_1_b, f2_re, f2_im);
  k_ifftH<<<NBATCH * CHM * NK, 128, 0, stream>>>(f2_re, f2_im, f1_re, f1_im);
  k_irfftW<<<NBATCH * CHM * 128, 128, 0, stream>>>(f1_re, f1_im, cat);

  // conv2 (WMMA) -> d_out
  k_conv2<<<NPIX / 128, TB, 0, stream>>>(cat, w2h, out);
  (void)in_sizes; (void)n_in; (void)out_size;
}
